// PointNetPlusSemseg_12824772346429
// MI455X (gfx1250) — matrix operand order repaired and checked by reading, compiled, not run on hardware
//
#include <hip/hip_runtime.h>
#include <hip/hip_bf16.h>

typedef __attribute__((ext_vector_type(16))) _Float16 v16h;
typedef __attribute__((ext_vector_type(8)))  _Float16 v8h;
typedef __attribute__((ext_vector_type(8)))  float    v8f;

#define CEILDIV(a,b) (((a)+(b)-1)/(b))
static inline int r32i(int v) { return (v + 31) & ~31; }
static inline int r64i(int v) { return (v + 63) & ~63; }

// ---------------------------------------------------------------------------
// Elementwise helpers
// ---------------------------------------------------------------------------
__global__ void k_copy_strided(const _Float16* __restrict__ in, _Float16* __restrict__ out,
                               int total, int C, int stride) {
  int t = blockIdx.x * blockDim.x + threadIdx.x;
  if (t >= total) return;
  int row = t / C, c = t % C;
  out[t] = in[(size_t)row * stride + c];
}

// x[B,3,N] -> xyz[B,N,3]
__global__ void k_xyz_transpose(const float* __restrict__ x, float* __restrict__ xyz,
                                int Bn, int Nn) {
  int t = blockIdx.x * blockDim.x + threadIdx.x;
  int total = Bn * Nn;
  if (t >= total) return;
  int b = t / Nn, n = t % Nn;
  xyz[(size_t)t * 3 + 0] = x[((size_t)b * 3 + 0) * Nn + n];
  xyz[(size_t)t * 3 + 1] = x[((size_t)b * 3 + 1) * Nn + n];
  xyz[(size_t)t * 3 + 2] = x[((size_t)b * 3 + 2) * Nn + n];
}

// C[rows=B*N,13] f32 -> out[B,13,N]
__global__ void k_seg_transpose(const float* __restrict__ Cin, float* __restrict__ out,
                                int Bn, int Nn, int Cls) {
  int t = blockIdx.x * blockDim.x + threadIdx.x;
  int total = Bn * Cls * Nn;
  if (t >= total) return;
  int b = t / (Cls * Nn);
  int rem = t % (Cls * Nn);
  int o = rem / Nn, n = rem % Nn;
  out[t] = Cin[((size_t)b * Nn + n) * Cls + o];
}

// ---------------------------------------------------------------------------
// Weight pack: W[K,N] f32 (row-major) -> f16 fragment-order blocks.
// For col-tile ct (16 cols), k-block kb (32 rows), lane l (0..31): 16
// contiguous halves = B-fragment elements e=0..15, i.e. k = kb*32+(e>>3)*16+(l>>4)*8+(e&7),
// col = ct*16+(l&15). Zero-filled OOB (this carries ALL K/N bounds handling).
// ---------------------------------------------------------------------------
__global__ void k_pack_w(const float* __restrict__ W, _Float16* __restrict__ Wp,
                         int K, int N, int Kpad, int Npad) {
  int total = (Npad >> 4) * (Kpad >> 5) * 512;
  int t = blockIdx.x * blockDim.x + threadIdx.x;
  if (t >= total) return;
  int e = t & 15;
  int lane = (t >> 4) & 31;
  int blk = t >> 9;
  int nk = Kpad >> 5;
  int ct = blk / nk, kb = blk % nk;
  int col = ct * 16 + (lane & 15);
  int k = kb * 32 + (e >> 3) * 16 + (lane >> 4) * 8 + (e & 7);
  float v = (k < K && col < N) ? W[(size_t)k * N + col] : 0.f;
  Wp[t] = (_Float16)v;
}

// ---------------------------------------------------------------------------
// Farthest point sampling: one 1024-thread workgroup per batch; running
// min-dist array in LDS (32KB of the 320KB WGP pool) + LDS tree argmax.
// ---------------------------------------------------------------------------
__global__ __launch_bounds__(1024) void k_fps(const float* __restrict__ xyz,
                                              float* __restrict__ new_xyz,
                                              int Nn, int S) {
  __shared__ float dist[8192];
  __shared__ float rv[1024];
  __shared__ int   ri[1024];
  int b = blockIdx.x;
  int tid = threadIdx.x;
  const float* xb = xyz + (size_t)b * Nn * 3;
  for (int j = tid; j < Nn; j += blockDim.x) dist[j] = 1e10f;
  __syncthreads();
  int far = 0;
  for (int it = 0; it < S; ++it) {
    float cx = xb[far * 3 + 0], cy = xb[far * 3 + 1], cz = xb[far * 3 + 2];
    if (tid == 0) {
      new_xyz[((size_t)b * S + it) * 3 + 0] = cx;
      new_xyz[((size_t)b * S + it) * 3 + 1] = cy;
      new_xyz[((size_t)b * S + it) * 3 + 2] = cz;
    }
    float bestv = -1.0f; int besti = Nn - 1;
    for (int j = tid; j < Nn; j += blockDim.x) {
      float dx = xb[j * 3 + 0] - cx;
      float dy = xb[j * 3 + 1] - cy;
      float dz = xb[j * 3 + 2] - cz;
      float d = dx * dx + dy * dy + dz * dz;
      float dm = fminf(dist[j], d);
      dist[j] = dm;
      if (dm > bestv || (dm == bestv && j < besti)) { bestv = dm; besti = j; }
    }
    rv[tid] = bestv; ri[tid] = besti;
    __syncthreads();
    for (int off = blockDim.x >> 1; off > 0; off >>= 1) {
      if (tid < off) {
        float v2 = rv[tid + off]; int i2 = ri[tid + off];
        if (v2 > rv[tid] || (v2 == rv[tid] && i2 < ri[tid])) { rv[tid] = v2; ri[tid] = i2; }
      }
      __syncthreads();
    }
    far = ri[0];
    __syncthreads();
  }
}

// ---------------------------------------------------------------------------
// Ball query: one thread per (b,s). First k ascending indices within radius,
// padded with the first hit (matches sort+slice+where in the reference).
// ---------------------------------------------------------------------------
__global__ void k_ball_query(const float* __restrict__ xyz, const float* __restrict__ new_xyz,
                             int* __restrict__ gidx, int Nn, int S, float r2, int total) {
  int t = blockIdx.x * blockDim.x + threadIdx.x;
  if (t >= total) return;
  int b = t / S;
  const float* xb = xyz + (size_t)b * Nn * 3;
  const float* c = new_xyz + (size_t)t * 3;
  float cx = c[0], cy = c[1], cz = c[2];
  int* out = gidx + (size_t)t * 32;
  int cnt = 0; int first = 0;
  for (int j = 0; j < Nn && cnt < 32; ++j) {
    float dx = xb[j * 3 + 0] - cx;
    float dy = xb[j * 3 + 1] - cy;
    float dz = xb[j * 3 + 2] - cz;
    float d = dx * dx + dy * dy + dz * dz;
    if (d <= r2) { if (cnt == 0) first = j; out[cnt++] = j; }
  }
  for (; cnt < 32; ++cnt) out[cnt] = first;
}

// ---------------------------------------------------------------------------
// Group + center + concat feats -> f16 GEMM A (row stride astride, pad cols
// zeroed so the WMMA A-side never needs bounds checks).
// ---------------------------------------------------------------------------
__global__ void k_group(const float* __restrict__ xyz, const float* __restrict__ new_xyz,
                        const _Float16* __restrict__ feats, const int* __restrict__ gidx,
                        _Float16* __restrict__ Aout, int total, int S, int Nn, int Cf,
                        int astride) {
  int row = blockIdx.x * blockDim.x + threadIdx.x;
  if (row >= total) return;
  int b = row / (S * 32);
  int rem = row % (S * 32);
  int s = rem / 32;
  int gi = gidx[row];
  const float* p = xyz + ((size_t)b * Nn + gi) * 3;
  const float* c = new_xyz + ((size_t)b * S + s) * 3;
  int cin = 3 + Cf;
  _Float16* a = Aout + (size_t)row * astride;
  a[0] = (_Float16)(p[0] - c[0]);
  a[1] = (_Float16)(p[1] - c[1]);
  a[2] = (_Float16)(p[2] - c[2]);
  if (Cf > 0) {
    const _Float16* f = feats + ((size_t)b * Nn + gi) * Cf;
    for (int cc = 0; cc < Cf; ++cc) a[3 + cc] = f[cc];
  }
  for (int cc = cin; cc < astride; ++cc) a[cc] = (_Float16)0.f;
}

// ---------------------------------------------------------------------------
// Max over k=32 neighbors: h[B*S*32, stride] f16 -> feats[B*S, C] f16 (dense).
// ---------------------------------------------------------------------------
__global__ void k_maxpool(const _Float16* __restrict__ h, _Float16* __restrict__ feats,
                          int BS, int C, int stride) {
  int t = blockIdx.x * blockDim.x + threadIdx.x;
  if (t >= BS * C) return;
  int bs = t / C, c = t % C;
  const _Float16* hp = h + (size_t)bs * 32 * stride + c;
  float m = -3.0e38f;
  for (int kk = 0; kk < 32; ++kk) m = fmaxf(m, (float)hp[(size_t)kk * stride]);
  feats[t] = (_Float16)m;
}

// ---------------------------------------------------------------------------
// FP stage: 3-NN inverse-distance interp + concat target feats -> f16 GEMM A.
// ---------------------------------------------------------------------------
__global__ void k_fp_build(const float* __restrict__ xyz_src, const _Float16* __restrict__ fsrc,
                           int Ss, int Csrc,
                           const float* __restrict__ xyz_tgt, const _Float16* __restrict__ ftgt,
                           int St, int Ctgt,
                           _Float16* __restrict__ Aout, int total, int astride) {
  int t = blockIdx.x * blockDim.x + threadIdx.x;
  if (t >= total) return;
  int b = t / St, tt = t % St;
  const float* pt = xyz_tgt + ((size_t)b * St + tt) * 3;
  float px = pt[0], py = pt[1], pz = pt[2];
  float d0 = 3.4e38f, d1 = 3.4e38f, d2 = 3.4e38f;
  int i0 = 0, i1 = 0, i2 = 0;
  const float* xs = xyz_src + (size_t)b * Ss * 3;
  for (int s = 0; s < Ss; ++s) {
    float dx = xs[s * 3 + 0] - px;
    float dy = xs[s * 3 + 1] - py;
    float dz = xs[s * 3 + 2] - pz;
    float d = dx * dx + dy * dy + dz * dz;
    if (d < d0)      { d2 = d1; i2 = i1; d1 = d0; i1 = i0; d0 = d; i0 = s; }
    else if (d < d1) { d2 = d1; i2 = i1; d1 = d;  i1 = s; }
    else if (d < d2) { d2 = d;  i2 = s; }
  }
  float w0 = 1.0f / (d0 + 1e-8f);
  float w1 = 1.0f / (d1 + 1e-8f);
  float w2 = 1.0f / (d2 + 1e-8f);
  float wsum = w0 + w1 + w2;
  w0 /= wsum; w1 /= wsum; w2 /= wsum;
  const _Float16* f0 = fsrc + ((size_t)b * Ss + i0) * Csrc;
  const _Float16* f1 = fsrc + ((size_t)b * Ss + i1) * Csrc;
  const _Float16* f2 = fsrc + ((size_t)b * Ss + i2) * Csrc;
  int cin = Csrc + Ctgt;
  _Float16* a = Aout + (size_t)t * astride;
  for (int c = 0; c < Csrc; ++c)
    a[c] = (_Float16)(w0 * (float)f0[c] + w1 * (float)f1[c] + w2 * (float)f2[c]);
  if (Ctgt > 0) {
    const _Float16* ft = ftgt + ((size_t)b * St + tt) * Ctgt;
    for (int c = 0; c < Ctgt; ++c) a[Csrc + c] = ft[c];
  }
  for (int c = cin; c < astride; ++c) a[c] = (_Float16)0.f;
}

// ---------------------------------------------------------------------------
// Fragment loaders (unconditional 16B vector loads, ISA 7.12.2 layouts).
// ---------------------------------------------------------------------------
__device__ __forceinline__ v16h ld_afrag(const _Float16* p) {
  // A 16x32 f16: lane {m,lh}: halves [lh*8, lh*8+8) and [16+lh*8, ...+8)
  v8h x = *(const v8h*)p;
  v8h y = *(const v8h*)(p + 16);
  return __builtin_shufflevector(x, y, 0,1,2,3,4,5,6,7,8,9,10,11,12,13,14,15);
}
__device__ __forceinline__ v16h ld_bfrag(const _Float16* p) {
  // packed per-lane fragment order: 16 contiguous halves
  v8h x = *(const v8h*)p;
  v8h y = *(const v8h*)(p + 8);
  return __builtin_shufflevector(x, y, 0,1,2,3,4,5,6,7,8,9,10,11,12,13,14,15);
}

// ---------------------------------------------------------------------------
// WMMA GEMM: D = relu(A x W + bias), f32 accumulate.
// Register-blocked: each wave computes a 32x64 tile (2 row-frags x 4 col-frags
// = 8 accumulators); per K-step it loads 2 A-frags + 4 B-frags (6 KB) and
// issues 8 WMMAs (~11 FLOP/byte from L2 vs ~4 unblocked). 8 waves / 256
// threads -> block covers 256 rows x 64 cols. All row counts in this network
// are multiples of 256; N padded to 64 in packed weights (zero-filled), so
// pad tiles compute exact zeros and stores are guarded only by ostride.
// ---------------------------------------------------------------------------
__global__ __launch_bounds__(256) void k_gemm(const _Float16* __restrict__ A,
                                              const _Float16* __restrict__ Wp,
                                              const float* __restrict__ bias,
                                              _Float16* __restrict__ out16,
                                              float* __restrict__ out32,
                                              int Kpad, int N, int astride,
                                              int ostride, int relu) {
  const int wave = threadIdx.x >> 5;
  const int lane = threadIdx.x & 31;
  const int m  = lane & 15;
  const int lh = lane >> 4;
  const int row0 = blockIdx.y * 256 + wave * 32;
  const int col0 = blockIdx.x * 64;
  const int nk = Kpad >> 5;
  const size_t ts = (size_t)nk * 512;   // halves per 16-col weight tile

  const _Float16* ar0 = A + (size_t)(row0 + m) * astride + lh * 8;
  const _Float16* ar1 = ar0 + (size_t)16 * astride;
  const _Float16* bp0 = Wp + (size_t)blockIdx.x * 4 * ts + lane * 16;
  const _Float16* bp1 = bp0 + ts;
  const _Float16* bp2 = bp1 + ts;
  const _Float16* bp3 = bp2 + ts;

  v8f c00 = {}, c01 = {}, c02 = {}, c03 = {};
  v8f c10 = {}, c11 = {}, c12 = {}, c13 = {};

#define WMMA16(af, bf, cc) \
  __builtin_amdgcn_wmma_f32_16x16x32_f16(false, (af), false, (bf), (short)0, (cc), false, false)

  for (int kb = 0; kb < Kpad; kb += 32) {
    v16h a0 = ld_afrag(ar0 + kb);
    v16h a1 = ld_afrag(ar1 + kb);
    v16h b0 = ld_bfrag(bp0);
    v16h b1 = ld_bfrag(bp1);
    v16h b2 = ld_bfrag(bp2);
    v16h b3 = ld_bfrag(bp3);
    bp0 += 512; bp1 += 512; bp2 += 512; bp3 += 512;
    if (kb + 32 < Kpad) {
      __builtin_prefetch(ar0 + kb + 32, 0, 0);
      __builtin_prefetch(ar1 + kb + 32, 0, 0);
    }
    c00 = WMMA16(a0, b0, c00);
    c10 = WMMA16(a1, b0, c10);
    c01 = WMMA16(a0, b1, c01);
    c11 = WMMA16(a1, b1, c11);
    c02 = WMMA16(a0, b2, c02);
    c12 = WMMA16(a1, b2, c12);
    c03 = WMMA16(a0, b3, c03);
    c13 = WMMA16(a1, b3, c13);
  }
#undef WMMA16

  v8f accs[2][4] = {{c00, c01, c02, c03}, {c10, c11, c12, c13}};
#pragma unroll
  for (int rt = 0; rt < 2; ++rt) {
#pragma unroll
    for (int ct = 0; ct < 4; ++ct) {
      int ocol = col0 + ct * 16 + m;
      if (ocol >= ostride) continue;
      float bv = (ocol < N) ? bias[ocol] : 0.f;
#pragma unroll
      for (int i = 0; i < 8; ++i) {
        int orow = row0 + rt * 16 + lh * 8 + i;
        float v = accs[rt][ct][i] + bv;
        if (relu) v = fmaxf(v, 0.f);
        if (out16) out16[(size_t)orow * ostride + ocol] = (_Float16)v;
        else       out32[(size_t)orow * ostride + ocol] = v;
      }
    }
  }
}

// ---------------------------------------------------------------------------
// Host side
// ---------------------------------------------------------------------------
static inline void launch_gemm(const _Float16* A, const _Float16* Wp, const float* bias,
                               _Float16* o16, float* o32, int rows, int K, int N,
                               int ostride, int relu, hipStream_t st) {
  int Kpad = r32i(K), Npad = r64i(N);
  dim3 g(Npad / 64, rows / 256);
  k_gemm<<<g, 256, 0, st>>>(A, Wp, bias, o16, o32, Kpad, N, Kpad, ostride, relu);
}

extern "C" void kernel_launch(void* const* d_in, const int* in_sizes, int n_in,
                              void* d_out, int out_size, void* d_ws, size_t ws_size,
                              hipStream_t stream) {
  (void)in_sizes; (void)n_in; (void)out_size; (void)ws_size;
  constexpr int Bn = 4, Nn = 8192, CLS = 13;
  const int SL[5]   = {8192, 1024, 256, 64, 16};
  const float R2[4] = {0.01f, 0.04f, 0.16f, 0.64f};
  const int sa_ch[4][4] = {{3,32,48,64},{67,64,96,128},{131,128,196,256},{259,256,384,512}};
  const int fp_ch[4][4] = {{768,1024,512,0},{640,512,512,0},{576,512,256,0},{256,256,256,128}};
  const int fp_nl[4] = {2, 2, 2, 3};

  // ---- parse inputs -------------------------------------------------------
  int p = 0;
  const float* x = (const float*)d_in[p++];
  const float* saW[4][3]; const float* saB[4][3];
  for (int s = 0; s < 4; ++s)
    for (int l = 0; l < 3; ++l) { saW[s][l] = (const float*)d_in[p++]; saB[s][l] = (const float*)d_in[p++]; }
  const float* fpW[4][3]; const float* fpB[4][3];
  for (int i = 0; i < 4; ++i)
    for (int l = 0; l < fp_nl[i]; ++l) { fpW[i][l] = (const float*)d_in[p++]; fpB[i][l] = (const float*)d_in[p++]; }
  const float* headW = (const float*)d_in[p++];
  const float* headB = (const float*)d_in[p++];

  // ---- workspace bump allocator ------------------------------------------
  char* base = (char*)d_ws;
  size_t off = 0;
  auto alloc = [&](size_t bytes) -> void* {
    void* r = base + off;
    off = (off + bytes + 255) & ~(size_t)255;
    return r;
  };
  auto pack = [&](const float* W, int K, int N) -> _Float16* {
    int Kpad = r32i(K), Npad = r64i(N);
    int total = (Npad >> 4) * (Kpad >> 5) * 512;
    _Float16* Wp = (_Float16*)alloc((size_t)total * 2);
    k_pack_w<<<CEILDIV(total, 256), 256, 0, stream>>>(W, Wp, K, N, Kpad, Npad);
    return Wp;
  };

  // packed f16 weights (zero-filled to Kpad x Npad)
  _Float16* saWp[4][3];
  for (int s = 0; s < 4; ++s)
    for (int l = 0; l < 3; ++l)
      saWp[s][l] = pack(saW[s][l], sa_ch[s][l], sa_ch[s][l + 1]);
  _Float16* fpWp[4][3];
  for (int i = 0; i < 4; ++i)
    for (int l = 0; l < fp_nl[i]; ++l)
      fpWp[i][l] = pack(fpW[i][l], fp_ch[i][l], fp_ch[i][l + 1]);
  _Float16* headWp = pack(headW, 128, CLS);

  // activation / scratch buffers
  const size_t HBUF = (size_t)9 * 1024 * 1024;  // f16 elems (covers 131072x64 & 32768x256)
  _Float16* Abuf = (_Float16*)alloc(HBUF * 2);
  _Float16* Hbuf = (_Float16*)alloc(HBUF * 2);
  int*   gidx = (int*)alloc((size_t)Bn * 1024 * 32 * 4);
  float* Cbuf = (float*)alloc((size_t)Bn * Nn * CLS * 4);

  float* xyzL[5];
  _Float16* featsL[5];
  int CfL[5];
  xyzL[0] = (float*)alloc((size_t)Bn * Nn * 3 * 4);
  featsL[0] = nullptr; CfL[0] = 0;
  k_xyz_transpose<<<CEILDIV(Bn * Nn, 256), 256, 0, stream>>>(x, xyzL[0], Bn, Nn);

  // ---- SA (set abstraction) stages ---------------------------------------
  for (int s = 0; s < 4; ++s) {
    int Np = SL[s], Sn = SL[s + 1], Cf = CfL[s];
    xyzL[s + 1] = (float*)alloc((size_t)Bn * Sn * 3 * 4);
    k_fps<<<Bn, 1024, 0, stream>>>(xyzL[s], xyzL[s + 1], Np, Sn);
    int tq = Bn * Sn;
    k_ball_query<<<CEILDIV(tq, 256), 256, 0, stream>>>(xyzL[s], xyzL[s + 1], gidx, Np, Sn, R2[s], tq);
    int rows = Bn * Sn * 32;
    int cin = 3 + Cf;
    k_group<<<CEILDIV(rows, 256), 256, 0, stream>>>(xyzL[s], xyzL[s + 1], featsL[s], gidx,
                                                    Abuf, rows, Sn, Np, Cf, r32i(cin));
    _Float16* cur = Abuf; _Float16* nxt = Hbuf;
    for (int l = 0; l < 3; ++l) {
      launch_gemm(cur, saWp[s][l], saB[s][l], nxt, nullptr,
                  rows, sa_ch[s][l], sa_ch[s][l + 1], r32i(sa_ch[s][l + 1]), 1, stream);
      _Float16* tmp = cur; cur = nxt; nxt = tmp;
    }
    int Co = sa_ch[s][3];
    featsL[s + 1] = (_Float16*)alloc((size_t)Bn * Sn * Co * 2);
    k_maxpool<<<CEILDIV(Bn * Sn * Co, 256), 256, 0, stream>>>(cur, featsL[s + 1],
                                                              Bn * Sn, Co, r32i(Co));
    CfL[s + 1] = Co;
  }

  // ---- FP (feature propagation) stages -----------------------------------
  const _Float16* fsrc = featsL[4];
  int Csrc = CfL[4];
  for (int i = 0; i < 4; ++i) {
    int srcL = 4 - i, tgtL = 3 - i;
    int Ss = SL[srcL], St = SL[tgtL];
    int Ctgt = CfL[tgtL];
    int rows = Bn * St;
    int cin = Csrc + Ctgt;
    k_fp_build<<<CEILDIV(rows, 256), 256, 0, stream>>>(xyzL[srcL], fsrc, Ss, Csrc,
                                                       xyzL[tgtL], featsL[tgtL], St, Ctgt,
                                                       Abuf, rows, r32i(cin));
    _Float16* cur = Abuf; _Float16* nxt = Hbuf;
    for (int l = 0; l < fp_nl[i]; ++l) {
      launch_gemm(cur, fpWp[i][l], fpB[i][l], nxt, nullptr,
                  rows, fp_ch[i][l], fp_ch[i][l + 1], r32i(fp_ch[i][l + 1]), 1, stream);
      _Float16* tmp = cur; cur = nxt; nxt = tmp;
    }
    int Co = fp_ch[i][fp_nl[i]];
    if (i < 3) {
      _Float16* keep = (_Float16*)alloc((size_t)rows * Co * 2);
      k_copy_strided<<<CEILDIV(rows * Co, 256), 256, 0, stream>>>(cur, keep, rows * Co,
                                                                  Co, r32i(Co));
      fsrc = keep; Csrc = Co;
    } else {
      // segmentation head: (B*N,128) x (128,13) + bias, no relu, f32 out (stride 13)
      launch_gemm(cur, headWp, headB, nullptr, Cbuf, rows, 128, CLS, CLS, 0, stream);
      k_seg_transpose<<<CEILDIV(Bn * CLS * Nn, 256), 256, 0, stream>>>(Cbuf, (float*)d_out,
                                                                       Bn, Nn, CLS);
    }
  }
}
